// Mamba2_35356170780768
// MI455X (gfx1250) — matrix operand order repaired and checked by reading, not compiled, not run on hardware
//
#include <hip/hip_runtime.h>
#include <math.h>

// ---------------- model constants ----------------
#define D_MODEL   1024
#define D_STATE   128
#define D_CONV    4
#define HEADDIM   64
#define D_INNER   2048
#define D_SSM     2048
#define NHEADS    32          // D_SSM / HEADDIM
#define CONV_DIM  2304        // D_SSM + 2*D_STATE
#define D_IN_PROJ 4384        // 2*D_INNER + 2*D_STATE + NHEADS
#define BATCH     2
#define SEQLEN    4096
#define ROWS      (BATCH * SEQLEN)   // 8192

typedef __attribute__((ext_vector_type(16))) __bf16 v16bf;
typedef __attribute__((ext_vector_type(8)))  float  v8f;

union Frag {
    v16bf v;
    float4 q[2];   // 2 x 16B contiguous chunks (8 bf16 each)
};

struct FragSet {
    Frag a[2];     // two 16x32 A tiles (32 M-rows)
    Frag b[4];     // four 32x16 B tiles (64 N-cols)
};

// ---------------- f32 -> bf16 convert ----------------
__global__ __launch_bounds__(256) void f32_to_bf16_kernel(const float* __restrict__ in,
                                                          __bf16* __restrict__ out,
                                                          unsigned int n) {
    unsigned int i = blockIdx.x * 256u + threadIdx.x;
    if (i < n) out[i] = (__bf16)in[i];
}

// ---------------- WMMA bf16 GEMM: C[M,N] = A[M,K] * B[N,K]^T ----------------
// Block = 256 threads = 8 waves, tiled 2(M) x 4(N); each wave computes a
// 32x64 strip of C (2x4 WMMA tiles). Fragment sets are double-buffered so the
// 12 b128 loads of the next K-step stay in flight under the 8 WMMAs of the
// current K-step (compiler can wait loadcnt<=12 instead of draining to 0).
// Requires: M % 64 == 0, K % 64 == 0. N may be ragged (clamped loads, guarded stores).
__global__ __launch_bounds__(256) void wmma_gemm_bf16_nt(const __bf16* __restrict__ A,
                                                         const __bf16* __restrict__ B,
                                                         float* __restrict__ C,
                                                         int M, int N, int K) {
    const int tid  = threadIdx.x;
    const int wave = tid >> 5;
    const int lane = tid & 31;
    const int hi   = lane >> 4;     // half-wave select (ISA A/B fragment layout)
    const int lm   = lane & 15;

    const int wm = wave & 1;        // 2 waves along M
    const int wn = wave >> 1;       // 4 waves along N
    const int tm_base = blockIdx.y * 4 + wm * 2;          // first 16-row tile (wave owns 2)
    const int tn_base = (blockIdx.x * 4 + wn) * 4;        // first 16-col tile (wave owns 4)

    const __bf16* Arow0 = A + (size_t)(tm_base * 16 + lm) * K;
    const __bf16* Arow1 = Arow0 + (size_t)16 * K;

    // B rows for the 4 n-tiles (clamped so loads stay in-bounds on ragged N)
    const __bf16* Brow0;
    const __bf16* Brow1;
    const __bf16* Brow2;
    const __bf16* Brow3;
    {
        int n0 = (tn_base + 0) * 16 + lm; if (n0 >= N) n0 = N - 1;
        int n1 = (tn_base + 1) * 16 + lm; if (n1 >= N) n1 = N - 1;
        int n2 = (tn_base + 2) * 16 + lm; if (n2 >= N) n2 = N - 1;
        int n3 = (tn_base + 3) * 16 + lm; if (n3 >= N) n3 = N - 1;
        Brow0 = B + (size_t)n0 * K;
        Brow1 = B + (size_t)n1 * K;
        Brow2 = B + (size_t)n2 * K;
        Brow3 = B + (size_t)n3 * K;
    }

    v8f acc[2][4];
#pragma unroll
    for (int r = 0; r < 2; ++r)
#pragma unroll
        for (int t = 0; t < 4; ++t)
            acc[r][t] = (v8f){0.f,0.f,0.f,0.f,0.f,0.f,0.f,0.f};

    // A fragment (16x32): lane = row, lo lanes K k..k+15, hi lanes k+16..k+31.
    // B fragment (32x16): lane = column, lo lanes K k..k+15, hi lanes k+16..k+31.
#define LOAD_SET(S, kk)                                                        \
    do {                                                                       \
        (S).a[0].q[0] = *(const float4*)(Arow0 + (kk) + hi * 16);              \
        (S).a[0].q[1] = *(const float4*)(Arow0 + (kk) + hi * 16 + 8);          \
        (S).a[1].q[0] = *(const float4*)(Arow1 + (kk) + hi * 16);              \
        (S).a[1].q[1] = *(const float4*)(Arow1 + (kk) + hi * 16 + 8);          \
        (S).b[0].q[0] = *(const float4*)(Brow0 + (kk) + hi * 16);              \
        (S).b[0].q[1] = *(const float4*)(Brow0 + (kk) + hi * 16 + 8);          \
        (S).b[1].q[0] = *(const float4*)(Brow1 + (kk) + hi * 16);              \
        (S).b[1].q[1] = *(const float4*)(Brow1 + (kk) + hi * 16 + 8);          \
        (S).b[2].q[0] = *(const float4*)(Brow2 + (kk) + hi * 16);              \
        (S).b[2].q[1] = *(const float4*)(Brow2 + (kk) + hi * 16 + 8);          \
        (S).b[3].q[0] = *(const float4*)(Brow3 + (kk) + hi * 16);              \
        (S).b[3].q[1] = *(const float4*)(Brow3 + (kk) + hi * 16 + 8);          \
    } while (0)

#define MMA_SET(S)                                                             \
    do {                                                                       \
        _Pragma("unroll")                                                      \
        for (int r = 0; r < 2; ++r) {                                          \
            _Pragma("unroll")                                                  \
            for (int t = 0; t < 4; ++t) {                                      \
                acc[r][t] = __builtin_amdgcn_wmma_f32_16x16x32_bf16(           \
                    false, (S).a[r].v, false, (S).b[t].v,                      \
                    (short)0, acc[r][t], false, false);                        \
            }                                                                  \
        }                                                                      \
    } while (0)

    FragSet s0, s1;
    LOAD_SET(s0, 0);
    LOAD_SET(s1, 32);
    int k = 0;
    for (; k + 64 < K; k += 64) {
        __builtin_prefetch(Arow0 + k + 128, 0, 3);
        MMA_SET(s0);
        LOAD_SET(s0, k + 64);
        MMA_SET(s1);
        LOAD_SET(s1, k + 96);
    }
    MMA_SET(s0);
    MMA_SET(s1);

#undef LOAD_SET
#undef MMA_SET

    // C/D layout: VGPR v -> row (v + hi*8), lanes give column
#pragma unroll
    for (int r = 0; r < 2; ++r) {
#pragma unroll
        for (int t = 0; t < 4; ++t) {
            const int n = (tn_base + t) * 16 + lm;
            if (n < N) {
                float* Cb = C + (size_t)((tm_base + r) * 16 + hi * 8) * N + n;
#pragma unroll
                for (int v = 0; v < 8; ++v) Cb[(size_t)v * N] = acc[r][t][v];
            }
        }
    }
}

// ---------------- depthwise causal conv (4 taps) + SiLU ----------------
__global__ __launch_bounds__(256) void conv_silu_kernel(const float* __restrict__ zx,
                                                        const float* __restrict__ cw,
                                                        const float* __restrict__ cb,
                                                        float* __restrict__ xbc) {
    const unsigned int idx = blockIdx.x * 256u + threadIdx.x;
    const unsigned int total = (unsigned int)ROWS * CONV_DIM;
    if (idx >= total) return;
    const int c  = idx % CONV_DIM;
    const int rl = idx / CONV_DIM;          // b*L + l
    const int l  = rl % SEQLEN;

    float acc = cb[c];
#pragma unroll
    for (int w = 0; w < D_CONV; ++w) {
        const int li = l + w - (D_CONV - 1);
        if (li >= 0)
            acc += zx[(size_t)(rl + li - l) * D_IN_PROJ + D_SSM + c] * cw[c * D_CONV + w];
    }
    acc = acc / (1.f + __expf(-acc));       // SiLU
    xbc[(size_t)rl * CONV_DIM + c] = acc;
}

// ---------------- dt softplus + dA = exp(dt * A) ----------------
__global__ __launch_bounds__(256) void dt_kernel(const float* __restrict__ zx,
                                                 const float* __restrict__ dt_bias,
                                                 const float* __restrict__ A_log,
                                                 float* __restrict__ dt_sp,
                                                 float* __restrict__ dAg) {
    const unsigned int idx = blockIdx.x * 256u + threadIdx.x;
    if (idx >= (unsigned int)ROWS * NHEADS) return;
    const int h = idx % NHEADS;
    const int r = idx / NHEADS;
    const float x  = zx[(size_t)r * D_IN_PROJ + D_SSM + CONV_DIM + h] + dt_bias[h];
    const float sp = (x > 20.f) ? x : log1pf(__expf(x));
    const float Ah = -__expf(A_log[h]);
    dt_sp[idx] = sp;
    dAg[idx]   = __expf(sp * Ah);
}

// ---------------- selective state scan ----------------
// One 1024-thread block per (b, head). State h[64 p x 128 n] lives in registers:
// thread = (p = tid>>4, j = tid&15) owns n = j*8 .. j*8+7 (8 f32). Per step B/C/x
// staged in LDS; y reduced over the 16 j-lanes with wave32 shuffles.
__global__ __launch_bounds__(1024) void scan_kernel(const float* __restrict__ xBC,
                                                    const float* __restrict__ dt_sp,
                                                    const float* __restrict__ dAg,
                                                    const float* __restrict__ Dp,
                                                    float* __restrict__ y) {
    const int b = blockIdx.x >> 5;
    const int h = blockIdx.x & 31;
    const int tid = threadIdx.x;
    const int p  = tid >> 4;
    const int j  = tid & 15;
    const int n0 = j * 8;

    __shared__ float sB[D_STATE];
    __shared__ float sC[D_STATE];
    __shared__ float sx[HEADDIM];

    float hs[8];
#pragma unroll
    for (int i = 0; i < 8; ++i) hs[i] = 0.f;
    const float Dh = Dp[h];

    for (int l = 0; l < SEQLEN; ++l) {
        const size_t row = (size_t)b * SEQLEN + l;
        __syncthreads();   // protect LDS from previous step's readers
        if (tid < D_STATE)                     sB[tid]               = xBC[row * CONV_DIM + D_SSM + tid];
        else if (tid < 2 * D_STATE)            sC[tid - D_STATE]     = xBC[row * CONV_DIM + D_SSM + D_STATE + (tid - D_STATE)];
        else if (tid < 2 * D_STATE + HEADDIM)  sx[tid - 2 * D_STATE] = xBC[row * CONV_DIM + h * HEADDIM + (tid - 2 * D_STATE)];
        __syncthreads();

        const float dA  = dAg[row * NHEADS + h];
        const float dtx = dt_sp[row * NHEADS + h] * sx[p];
        float yp = 0.f;
#pragma unroll
        for (int i = 0; i < 8; ++i) {
            const float hn = hs[i] * dA + dtx * sB[n0 + i];
            hs[i] = hn;
            yp += hn * sC[n0 + i];
        }
        yp += __shfl_xor(yp, 1, 32);
        yp += __shfl_xor(yp, 2, 32);
        yp += __shfl_xor(yp, 4, 32);
        yp += __shfl_xor(yp, 8, 32);
        if (j == 0) y[row * D_SSM + h * HEADDIM + p] = yp + Dh * sx[p];
    }
}

// ---------------- gating (y * silu(z)), RMS norm, bf16 output ----------------
__global__ __launch_bounds__(256) void gate_norm_kernel(const float* __restrict__ y,
                                                        const float* __restrict__ zx,
                                                        const float* __restrict__ nw,
                                                        __bf16* __restrict__ g) {
    const int row = blockIdx.x;
    const int tid = threadIdx.x;
    float vals[8];
    float ss = 0.f;
#pragma unroll
    for (int i = 0; i < 8; ++i) {
        const int c = tid + i * 256;
        const float z  = zx[(size_t)row * D_IN_PROJ + c];
        const float gv = y[(size_t)row * D_SSM + c] * (z / (1.f + __expf(-z)));
        vals[i] = gv;
        ss += gv * gv;
    }
    for (int o = 16; o; o >>= 1) ss += __shfl_xor(ss, o, 32);
    __shared__ float sred[8];
    __shared__ float stot;
    if ((tid & 31) == 0) sred[tid >> 5] = ss;
    __syncthreads();
    if (tid == 0) {
        float t = 0.f;
        for (int i = 0; i < 8; ++i) t += sred[i];
        stot = rsqrtf(t / (float)D_SSM + 1e-5f);
    }
    __syncthreads();
    const float sc = stot;
#pragma unroll
    for (int i = 0; i < 8; ++i) {
        const int c = tid + i * 256;
        g[(size_t)row * D_SSM + c] = (__bf16)(vals[i] * sc * nw[c]);
    }
}

// ---------------- host-side orchestration ----------------
static inline size_t align256(size_t x) { return (x + 255) & ~(size_t)255; }

extern "C" void kernel_launch(void* const* d_in, const int* in_sizes, int n_in,
                              void* d_out, int out_size, void* d_ws, size_t ws_size,
                              hipStream_t stream) {
    const float* u       = (const float*)d_in[0];
    const float* W_in    = (const float*)d_in[1];
    const float* conv_w  = (const float*)d_in[2];
    const float* conv_b  = (const float*)d_in[3];
    const float* dt_bias = (const float*)d_in[4];
    const float* A_log   = (const float*)d_in[5];
    const float* D       = (const float*)d_in[6];
    const float* norm_w  = (const float*)d_in[7];
    const float* W_out   = (const float*)d_in[8];
    float* out = (float*)d_out;

    // workspace carve-out
    char* ws = (char*)d_ws;
    size_t off = 0;
    __bf16* u_bf   = (__bf16*)(ws + off); off = align256(off + (size_t)ROWS * D_MODEL * 2);
    __bf16* Wi_bf  = (__bf16*)(ws + off); off = align256(off + (size_t)D_IN_PROJ * D_MODEL * 2);
    __bf16* Wo_bf  = (__bf16*)(ws + off); off = align256(off + (size_t)D_MODEL * D_INNER * 2);
    float*  zx     = (float*)(ws + off);  off = align256(off + (size_t)ROWS * D_IN_PROJ * 4);
    float*  xbc    = (float*)(ws + off);  off = align256(off + (size_t)ROWS * CONV_DIM * 4);
    float*  dt_sp  = (float*)(ws + off);  off = align256(off + (size_t)ROWS * NHEADS * 4);
    float*  dAg    = (float*)(ws + off);  off = align256(off + (size_t)ROWS * NHEADS * 4);
    float*  ybuf   = (float*)(ws + off);  off = align256(off + (size_t)ROWS * D_SSM * 4);
    __bf16* g_bf   = (__bf16*)(ws + off); off = align256(off + (size_t)ROWS * D_SSM * 2);
    (void)ws_size; (void)n_in; (void)in_sizes; (void)out_size;

    // 1) f32 -> bf16 conversions for WMMA operands
    {
        unsigned int n;
        n = (unsigned int)ROWS * D_MODEL;
        f32_to_bf16_kernel<<<(n + 255) / 256, 256, 0, stream>>>(u, u_bf, n);
        n = (unsigned int)D_IN_PROJ * D_MODEL;
        f32_to_bf16_kernel<<<(n + 255) / 256, 256, 0, stream>>>(W_in, Wi_bf, n);
        n = (unsigned int)D_MODEL * D_INNER;
        f32_to_bf16_kernel<<<(n + 255) / 256, 256, 0, stream>>>(W_out, Wo_bf, n);
    }

    // 2) in_proj GEMM: zx[8192, 4384] = u[8192,1024] @ W_in[4384,1024]^T
    {
        dim3 grid((D_IN_PROJ + 255) / 256, ROWS / 64);
        wmma_gemm_bf16_nt<<<grid, 256, 0, stream>>>(u_bf, Wi_bf, zx, ROWS, D_IN_PROJ, D_MODEL);
    }

    // 3) depthwise causal conv + SiLU
    {
        const unsigned int n = (unsigned int)ROWS * CONV_DIM;
        conv_silu_kernel<<<(n + 255) / 256, 256, 0, stream>>>(zx, conv_w, conv_b, xbc);
    }

    // 4) dt softplus + dA
    {
        const unsigned int n = (unsigned int)ROWS * NHEADS;
        dt_kernel<<<(n + 255) / 256, 256, 0, stream>>>(zx, dt_bias, A_log, dt_sp, dAg);
    }

    // 5) selective scan: one block per (batch, head)
    scan_kernel<<<BATCH * NHEADS, 1024, 0, stream>>>(xbc, dt_sp, dAg, D, ybuf);

    // 6) gating + RMS norm -> bf16 g
    gate_norm_kernel<<<ROWS, 256, 0, stream>>>(ybuf, zx, norm_w, g_bf);

    // 7) out_proj GEMM: out[8192, 1024] = g[8192,2048] @ W_out[1024,2048]^T
    {
        dim3 grid((D_MODEL + 255) / 256, ROWS / 64);
        wmma_gemm_bf16_nt<<<grid, 256, 0, stream>>>(g_bf, Wo_bf, out, ROWS, D_MODEL, D_INNER);
    }
}
